// TransformerModel_27925877359095
// MI455X (gfx1250) — matrix operand order repaired and checked by reading, not compiled, not run on hardware
//
#include <hip/hip_runtime.h>

#define HH   128            // hidden
#define GG   512            // 4*H (gates)
#define TT   512            // sequence length
#define LL   128            // layers
#define FF   5              // input features
#define CHUNK 16            // timesteps per pipelined chunk
#define NCHUNK (TT / CHUNK) // 32

typedef __attribute__((ext_vector_type(16))) _Float16 v16h;
typedef __attribute__((ext_vector_type(8)))  _Float16 v8h;
typedef __attribute__((ext_vector_type(2)))  _Float16 v2h;
typedef __attribute__((ext_vector_type(8)))  float    v8f;
typedef __attribute__((ext_vector_type(4)))  uint32_t v4u;
typedef __attribute__((ext_vector_type(8)))  int      v8i;
typedef __attribute__((ext_vector_type(4)))  int      v4i;

#if __has_builtin(__builtin_amdgcn_tensor_load_to_lds)
#define USE_TDM 1
#else
#define USE_TDM 0
#endif
#if __has_builtin(__builtin_amdgcn_fdot2)
#define USE_FDOT2 1
#else
#define USE_FDOT2 0
#endif

// ---------------------------------------------------------------------------
// fp32 -> fp16 weight conversion (grid-stride)
// ---------------------------------------------------------------------------
__global__ void cvt_f32_to_f16(const float* __restrict__ in,
                               _Float16* __restrict__ out, int n) {
    for (int i = blockIdx.x * blockDim.x + threadIdx.x; i < n;
         i += gridDim.x * blockDim.x)
        out[i] = (_Float16)in[i];
}

// ---------------------------------------------------------------------------
// embedding: x[t][h] = src[t][:] . W[h][:] + b[h]   (written as fp16)
// ---------------------------------------------------------------------------
__global__ void embed_k(const float* __restrict__ src,
                        const float* __restrict__ W,
                        const float* __restrict__ b,
                        _Float16* __restrict__ out) {
    int idx = blockIdx.x * blockDim.x + threadIdx.x;
    if (idx >= TT * HH) return;
    int t = idx >> 7, h = idx & (HH - 1);
    float a = b[h];
#pragma unroll
    for (int f = 0; f < FF; ++f) a += src[t * FF + f] * W[h * FF + f];
    out[idx] = (_Float16)a;
}

#if USE_TDM
// ---------------------------------------------------------------------------
// 1-D Tensor-Data-Mover load: nbytes (multiple of 8) from global -> LDS.
// D# built per CDNA5 ISA ch.8: group0 {count=1, lds_addr, global_addr, type=2},
// group1 {data_size=8B, tensor_dim0=tile_dim0=n8, dim1=1, stride0=n8}.
// This toolchain exposes the 6-arg builtin (extra int32x8 group before cpol).
// ---------------------------------------------------------------------------
__device__ __forceinline__ void tdm_load_1d(uint32_t lds_off, const void* gsrc,
                                            uint32_t nbytes) {
    uint64_t ga = (uint64_t)(uintptr_t)gsrc;
    uint32_t n8 = nbytes >> 3;  // 8-byte elements (must fit in 16-bit tile_dim0)
    v4u g0 = { 1u,                                   // count=1, user load
               lds_off,                              // [63:32] lds_addr
               (uint32_t)ga,                         // [95:64] global_addr lo
               ((uint32_t)(ga >> 32) & 0x01FFFFFFu)  // [120:96] global_addr hi
                   | 0x80000000u };                  // [127:126] type=2
    v8i g1 = { (int)(3u << 16),                      // data_size=3 (8B), mask=0
               (int)((n8 & 0xFFFFu) << 16),          // tensor_dim0 lo16
               (int)((n8 >> 16) | (1u << 16)),       // tensor_dim0 hi | dim1=1
               (int)((n8 & 0xFFFFu) << 16),          // tile_dim0
               1,                                    // tile_dim1=1
               (int)n8,                              // tensor_dim0_stride lo32
               0, 0 };
    v4i z4 = { 0, 0, 0, 0 };
    v8i z8 = { 0, 0, 0, 0, 0, 0, 0, 0 };
    __builtin_amdgcn_tensor_load_to_lds(g0, g1, z4, z4, z8, 0);
}
#endif

// ---------------------------------------------------------------------------
// Persistent wavefront-pipelined LSTM stack. One workgroup == one layer.
// 256 threads (8 waves). Whole layer's weights live in LDS (fp16, 256KB),
// staged by the Tensor Data Mover. Input-GEMM per 16-step chunk via
// v_wmma_f32_16x16x32_f16; recurrence via v_dot2_f32_f16 from LDS.
// ---------------------------------------------------------------------------
__global__ __launch_bounds__(256, 1)
void lstm_pipeline(const _Float16* __restrict__ wih,   // [L][4H][H] fp16
                   const _Float16* __restrict__ whh,   // [L][4H][H] fp16
                   const float* __restrict__ bih,      // [L][4H]
                   const float* __restrict__ bhh,      // [L][4H]
                   _Float16* __restrict__ act,         // [(L+1)][T][H] fp16
                   int* __restrict__ progress)         // [L]
{
    extern __shared__ char smem_raw[];
    _Float16* wih_s  = (_Float16*)smem_raw;                 // 512*128 halves
    _Float16* whh_s  = wih_s + GG * HH;                     // 512*128 halves
    float*    pre_s  = (float*)(whh_s + GG * HH);           // 16*512 f32
    float*    bias_s = pre_s + CHUNK * GG;                  // 512 f32
    float*    h_s    = bias_s + GG;                         // 128 f32
    float*    g_s    = h_s + HH;                            // 512 f32
    _Float16* xs     = (_Float16*)(g_s + GG);               // 16*128 halves
    _Float16* h16_s  = xs + CHUNK * HH;                     // 128 halves

    const int l    = blockIdx.x;
    const int tid  = threadIdx.x;
    const int lane = tid & 31;
    const int wave = tid >> 5;      // 0..7
    const int hi   = lane >> 4;     // half-wave select
    const int lm   = lane & 15;

    // ---- stage this layer's weights into LDS ----
#if USE_TDM
    if (wave == 0) {
        tdm_load_1d((uint32_t)(uintptr_t)wih_s, wih + (size_t)l * GG * HH,
                    GG * HH * 2);
        tdm_load_1d((uint32_t)(uintptr_t)whh_s, whh + (size_t)l * GG * HH,
                    GG * HH * 2);
        __builtin_amdgcn_s_wait_tensorcnt(0);
    }
#else
    {
        const int n16 = (GG * HH) / 8;  // uint4 count = 8192
        const uint4* s0 = (const uint4*)(wih + (size_t)l * GG * HH);
        uint4* d0 = (uint4*)wih_s;
        for (int i = tid; i < n16; i += 256) d0[i] = s0[i];
        const uint4* s1 = (const uint4*)(whh + (size_t)l * GG * HH);
        uint4* d1 = (uint4*)whh_s;
        for (int i = tid; i < n16; i += 256) d1[i] = s1[i];
    }
#endif
    for (int j = tid; j < GG; j += 256) bias_s[j] = bih[l * GG + j] + bhh[l * GG + j];
    if (tid < HH) { h_s[tid] = 0.0f; h16_s[tid] = (_Float16)0.0f; }
    float c_state = 0.0f;           // cell state, owned by tid < 128
    __syncthreads();

    const _Float16* actin  = act + (size_t)l * TT * HH;
    _Float16*       actout = act + (size_t)(l + 1) * TT * HH;

    for (int cb = 0; cb < NCHUNK; ++cb) {
        // ---- wait for previous layer to publish this chunk ----
        if (l > 0) {
            if (tid == 0) {
                while (__hip_atomic_load(&progress[l - 1], __ATOMIC_ACQUIRE,
                                         __HIP_MEMORY_SCOPE_AGENT) <= cb)
                    __builtin_amdgcn_s_sleep(2);
            }
            __syncthreads();
        }
        if (cb + 1 < NCHUNK)
            __builtin_prefetch(actin + (size_t)(cb + 1) * CHUNK * HH, 0, 1);

        // ---- load X chunk (16x128 fp16) into LDS: 256 threads x 16B ----
        {
            const uint4* s = (const uint4*)(actin + (size_t)cb * CHUNK * HH);
            ((uint4*)xs)[tid] = s[tid];
            __syncthreads();
        }

        // ---- WMMA GEMM: pre[16][512] = X(16x128) @ Wih^T + (bih+bhh) ----
        {
            v8f acc[4] = {};
#pragma unroll
            for (int kc = 0; kc < HH; kc += 32) {
                // A fragment (16x32): lane m=lm, two contiguous 8-half runs
                const _Float16* ap = xs + lm * HH + kc + 8 * hi;
                v8h alo = *(const v8h*)ap;
                v8h ahi = *(const v8h*)(ap + 16);
                v16h a = __builtin_shufflevector(alo, ahi, 0, 1, 2, 3, 4, 5, 6, 7,
                                                 8, 9, 10, 11, 12, 13, 14, 15);
#pragma unroll
                for (int i = 0; i < 4; ++i) {
                    // B fragment (32x16): col n = row of Wih; two 8-half runs matching A
                    int n = (wave + 8 * i) * 16 + lm;
                    const _Float16* bp = wih_s + n * HH + kc + 8 * hi;
                    v8h blo = *(const v8h*)bp;
                    v8h bhi = *(const v8h*)(bp + 16);
                    v16h b = __builtin_shufflevector(blo, bhi, 0, 1, 2, 3, 4, 5, 6, 7,
                                                     8, 9, 10, 11, 12, 13, 14, 15);
                    acc[i] = __builtin_amdgcn_wmma_f32_16x16x32_f16(
                        false, a, false, b, (short)0, acc[i], false, false);
                }
            }
            // epilogue: C layout -> pre_s, fused bias add
#pragma unroll
            for (int i = 0; i < 4; ++i) {
                int col = (wave + 8 * i) * 16 + lm;
                float bv = bias_s[col];
#pragma unroll
                for (int r = 0; r < 8; ++r)
                    pre_s[(r + 8 * hi) * GG + col] = acc[i][r] + bv;
            }
        }
        __syncthreads();

        // ---- sequential recurrence over the 16 steps of this chunk ----
        for (int ts = 0; ts < CHUNK; ++ts) {
            const float* pr = pre_s + ts * GG;
            // gate preactivations: 512 outputs over 256 threads (2 each)
#pragma unroll
            for (int rep = 0; rep < 2; ++rep) {
                int j = tid + rep * 256;
                const _Float16* wr = whh_s + j * HH;
                float a = pr[j];
#if USE_FDOT2
#pragma unroll
                for (int k = 0; k < HH; k += 8) {
                    v8h w8 = *(const v8h*)(wr + k);
                    v8h x8 = *(const v8h*)(h16_s + k);
#pragma unroll
                    for (int q = 0; q < 4; ++q) {
                        v2h wp = { w8[2 * q], w8[2 * q + 1] };
                        v2h xp = { x8[2 * q], x8[2 * q + 1] };
                        a = __builtin_amdgcn_fdot2(wp, xp, a, false);
                    }
                }
#else
#pragma unroll
                for (int k = 0; k < HH; k += 8) {
                    v8h w8 = *(const v8h*)(wr + k);
#pragma unroll
                    for (int q = 0; q < 8; ++q)
                        a += (float)w8[q] * h_s[k + q];
                }
#endif
                g_s[j] = a;
            }
            __syncthreads();
            if (tid < HH) {
                float ig = 1.0f / (1.0f + __expf(-g_s[tid]));
                float fg = 1.0f / (1.0f + __expf(-g_s[tid + HH]));
                float gv = tanhf(g_s[tid + 2 * HH]);
                float og = 1.0f / (1.0f + __expf(-g_s[tid + 3 * HH]));
                c_state = fg * c_state + ig * gv;
                float hv = og * tanhf(c_state);
                h_s[tid] = hv;
                h16_s[tid] = (_Float16)hv;
                actout[(size_t)(cb * CHUNK + ts) * HH + tid] = (_Float16)hv;
            }
            __syncthreads();
        }

        // ---- publish chunk to next layer ----
        __threadfence();
        if (tid == 0)
            __hip_atomic_store(&progress[l], cb + 1, __ATOMIC_RELEASE,
                               __HIP_MEMORY_SCOPE_AGENT);
    }
}

// ---------------------------------------------------------------------------
// decoders: out[0:2] = 100*|h_p @ dec_W^T + dec_b|, out[2:8] = h_g @ decg^T + bg
// ---------------------------------------------------------------------------
__global__ void decode_k(const _Float16* __restrict__ hp,
                         const _Float16* __restrict__ hg,
                         const float* __restrict__ decW, const float* __restrict__ decb,
                         const float* __restrict__ decgW, const float* __restrict__ decgb,
                         float* __restrict__ out) {
    int tid = threadIdx.x;
    if (tid < 2) {
        float a = decb[tid];
        for (int k = 0; k < HH; ++k) a += decW[tid * HH + k] * (float)hp[k];
        out[tid] = 100.0f * fabsf(a);
    } else if (tid < 8) {
        int j = tid - 2;
        float a = decgb[j];
        for (int k = 0; k < HH; ++k) a += decgW[j * HH + k] * (float)hg[k];
        out[2 + j] = a;
    }
}

// ---------------------------------------------------------------------------
extern "C" void kernel_launch(void* const* d_in, const int* in_sizes, int n_in,
                              void* d_out, int out_size, void* d_ws, size_t ws_size,
                              hipStream_t stream) {
    (void)in_sizes; (void)n_in; (void)out_size; (void)ws_size;
    const float* src     = (const float*)d_in[0];
    const float* emb_W   = (const float*)d_in[1];
    const float* emb_b   = (const float*)d_in[2];
    const float* Wih     = (const float*)d_in[3];
    const float* Whh     = (const float*)d_in[4];
    const float* bih     = (const float*)d_in[5];
    const float* bhh     = (const float*)d_in[6];
    const float* dec_W   = (const float*)d_in[7];
    const float* dec_b   = (const float*)d_in[8];
    const float* emb_g_W = (const float*)d_in[9];
    const float* emb_g_b = (const float*)d_in[10];
    const float* Wih_g   = (const float*)d_in[11];
    const float* Whh_g   = (const float*)d_in[12];
    const float* bih_g   = (const float*)d_in[13];
    const float* bhh_g   = (const float*)d_in[14];
    const float* dec_g_W = (const float*)d_in[15];
    const float* dec_g_b = (const float*)d_in[16];
    float* out = (float*)d_out;

    // ---- workspace carve-out ----
    char* ws = (char*)d_ws;
    size_t off = 0;
    auto carve = [&](size_t bytes) -> void* {
        void* p = ws + off;
        off = (off + bytes + 255) & ~(size_t)255;
        return p;
    };
    const size_t WB  = (size_t)LL * GG * HH;       // weight elems per tensor
    const size_t ACT = (size_t)(LL + 1) * TT * HH; // activation elems per head
    _Float16* wih16_p = (_Float16*)carve(WB * 2);
    _Float16* whh16_p = (_Float16*)carve(WB * 2);
    _Float16* wih16_g = (_Float16*)carve(WB * 2);
    _Float16* whh16_g = (_Float16*)carve(WB * 2);
    _Float16* act_p   = (_Float16*)carve(ACT * 2);
    _Float16* act_g   = (_Float16*)carve(ACT * 2);
    int*      prog    = (int*)carve(2 * LL * sizeof(int));

    (void)hipMemsetAsync(prog, 0, 2 * LL * sizeof(int), stream);

    // fp32 -> fp16 weights
    cvt_f32_to_f16<<<4096, 256, 0, stream>>>(Wih,   wih16_p, (int)WB);
    cvt_f32_to_f16<<<4096, 256, 0, stream>>>(Whh,   whh16_p, (int)WB);
    cvt_f32_to_f16<<<4096, 256, 0, stream>>>(Wih_g, wih16_g, (int)WB);
    cvt_f32_to_f16<<<4096, 256, 0, stream>>>(Whh_g, whh16_g, (int)WB);

    // embeddings -> layer-0 activation buffers
    embed_k<<<(TT * HH + 255) / 256, 256, 0, stream>>>(src, emb_W, emb_b, act_p);
    embed_k<<<(TT * HH + 255) / 256, 256, 0, stream>>>(src, emb_g_W, emb_g_b, act_g);

    // dynamic LDS: Wih + Whh + pre + bias + h + g + xs + h16
    const size_t shmem = (size_t)GG * HH * 2 * 2      // 256 KB weights
                       + (size_t)CHUNK * GG * 4       // 32 KB pre
                       + (size_t)GG * 4               // bias
                       + (size_t)HH * 4               // h (f32)
                       + (size_t)GG * 4               // g
                       + (size_t)CHUNK * HH * 2       // xs
                       + (size_t)HH * 2;              // h (f16)
    (void)hipFuncSetAttribute((const void*)lstm_pipeline,
                              hipFuncAttributeMaxDynamicSharedMemorySize,
                              (int)shmem);

    lstm_pipeline<<<LL, 256, shmem, stream>>>(wih16_p, whh16_p, bih, bhh,
                                              act_p, prog);
    lstm_pipeline<<<LL, 256, shmem, stream>>>(wih16_g, whh16_g, bih_g, bhh_g,
                                              act_g, prog + LL);

    const _Float16* hp = act_p + (size_t)LL * TT * HH + (size_t)(TT - 1) * HH;
    const _Float16* hg = act_g + (size_t)LL * TT * HH + (size_t)(TT - 1) * HH;
    decode_k<<<1, 32, 0, stream>>>(hp, hg, dec_W, dec_b, dec_g_W, dec_g_b, out);
}
